// Attention_12635793785074
// MI455X (gfx1250) — matrix operand order repaired and checked by reading, not compiled, not run on hardware
//
#include <hip/hip_runtime.h>
#include <hip/hip_bf16.h>
#include <stdint.h>

// ---------------------------------------------------------------------------
// MI455X (gfx1250) attention block: bf16 WMMA GEMMs + flash attention with
// TDM (tensor_load_to_lds) double-buffered K/V staging shared by 8 waves.
// ---------------------------------------------------------------------------

typedef __bf16 bf16;
typedef bf16  v16bf __attribute__((ext_vector_type(16)));
typedef bf16  v8bf  __attribute__((ext_vector_type(8)));
typedef float v8f   __attribute__((ext_vector_type(8)));
typedef unsigned int v4u __attribute__((ext_vector_type(4)));
typedef int  v4i __attribute__((ext_vector_type(4)));
typedef int  v8i __attribute__((ext_vector_type(8)));

#define S_LEN   4096
#define DMODEL  2048
#define NHEADS  16
#define NKV     4
#define HD      128

#if __has_builtin(__builtin_amdgcn_tensor_load_to_lds)
#define HAVE_TDM 1
#else
#define HAVE_TDM 0
#endif

__device__ __forceinline__ bf16 f2bf(float f) {
  union { float f; uint32_t u; } a; a.f = f;
  uint32_t u = a.u + 0x7FFFu + ((a.u >> 16) & 1u);   // round-to-nearest-even
  union { uint16_t s; bf16 b; } o; o.s = (uint16_t)(u >> 16);
  return o.b;
}

union Frag16 { v16bf v; v8bf h[2]; };

#if HAVE_TDM
// 2D bf16 tile DMA: global (line length tensor_d0, line stride stride0 elems)
// -> LDS (tile_d0 x tile_d1 packed contiguously). D# per CDNA5 ISA ch.8.
__device__ __forceinline__ void tdm_load_2d(uint32_t lds_off, const void* gptr,
                                            uint32_t tensor_d0, uint32_t tensor_d1,
                                            uint32_t tile_d0, uint32_t tile_d1,
                                            uint64_t stride0_elems) {
  uint64_t ga = (uint64_t)(uintptr_t)gptr;
  v4u g0;
  g0.x = 1u;                                           // count=1, user D#
  g0.y = lds_off;                                      // lds_addr (bytes)
  g0.z = (uint32_t)ga;                                 // global_addr[31:0]
  g0.w = (uint32_t)((ga >> 32) & 0x01FFFFFFu) | (2u << 30);  // addr[56:32] | type=2
  v8i g1;
  g1[0] = (int)(1u << 16);                             // data_size=1 (2 bytes)
  g1[1] = (int)((tensor_d0 & 0xFFFFu) << 16);          // tensor_dim0[15:0]
  g1[2] = (int)(((tensor_d0 >> 16) & 0xFFFFu) | ((tensor_d1 & 0xFFFFu) << 16));
  g1[3] = (int)(((tensor_d1 >> 16) & 0xFFFFu) | ((tile_d0 & 0xFFFFu) << 16));
  g1[4] = (int)(tile_d1 & 0xFFFFu);                    // tile_dim1, tile_dim2=0
  g1[5] = (int)(uint32_t)(stride0_elems & 0xFFFFFFFFu);// tensor_dim0_stride lo
  g1[6] = (int)(uint32_t)((stride0_elems >> 32) & 0xFFFFu);
  g1[7] = 0;
  v4i z4 = {0, 0, 0, 0};
#if __clang_major__ >= 23
  v8i z8 = {0, 0, 0, 0, 0, 0, 0, 0};
  __builtin_amdgcn_tensor_load_to_lds(g0, g1, z4, z4, z8, 0);
#else
  __builtin_amdgcn_tensor_load_to_lds(g0, g1, z4, z4, 0);
#endif
}
__device__ __forceinline__ uint32_t lds_offset(const void* p) {
  return (uint32_t)(uintptr_t)p;   // LDS aperture: addr[31:0] = LDS byte offset
}
#endif

// --------------------------- cast / transpose ------------------------------

__global__ void cast_bf16_kernel(const float* __restrict__ src,
                                 bf16* __restrict__ dst, int n) {
  int i = blockIdx.x * blockDim.x + threadIdx.x;
  if (i < n) dst[i] = f2bf(src[i]);
}

// W[K,N] f32 -> Wt[N,K] bf16
__global__ void cast_transpose_kernel(const float* __restrict__ src,
                                      bf16* __restrict__ dst, int K, int N) {
  int i = blockIdx.x * blockDim.x + threadIdx.x;
  if (i >= K * N) return;
  int k = i / N, n = i % N;
  dst[(size_t)n * K + k] = f2bf(src[i]);
}

// vproj[S, NKV*HD] f32 -> Vt[NKV][HD][S] bf16
__global__ void v_transpose_kernel(const float* __restrict__ src,
                                   bf16* __restrict__ dst) {
  int i = blockIdx.x * blockDim.x + threadIdx.x;
  if (i >= S_LEN * NKV * HD) return;
  int s = i / (NKV * HD);
  int c = i - s * (NKV * HD);
  int kv = c / HD, d = c - kv * HD;
  dst[((size_t)kv * HD + d) * S_LEN + s] = f2bf(src[i]);
}

// ------------------------------ bf16 GEMM ----------------------------------
// C[M,N] f32 = A[M,K] bf16 (row major) x Bt[N,K] bf16 (B transposed).
// One wave -> 16x64 tile (4 accumulators), K stepped by 32.

__global__ void gemm_bf16_kernel(const bf16* __restrict__ A,
                                 const bf16* __restrict__ Bt,
                                 float* __restrict__ C,
                                 int M, int N, int K) {
  int wid  = (blockIdx.x * blockDim.x + threadIdx.x) >> 5;
  int lane = threadIdx.x & 31;
  int ntiles = N >> 6;
  int tm = wid / ntiles, tn = wid - tm * ntiles;
  if (tm * 16 >= M) return;

  int hi = lane >> 4, r15 = lane & 15;
  const bf16* arow = A + (size_t)(tm * 16 + r15) * K + hi * 16;

  v8f acc[4] = {};
  for (int k0 = 0; k0 < K; k0 += 32) {
    Frag16 a;
    a.h[0] = *(const v8bf*)(arow + k0);        // K = k0 + kbase .. +7
    a.h[1] = *(const v8bf*)(arow + k0 + 8);    // K = k0+8+kbase .. +7
#pragma unroll
    for (int j = 0; j < 4; ++j) {
      const bf16* bp = Bt + (size_t)(tn * 64 + j * 16 + r15) * K + k0 + hi * 16;
      v16bf b = *(const v16bf*)bp;
      acc[j] = __builtin_amdgcn_wmma_f32_16x16x32_bf16(
          false, a.v, false, b, (short)0, acc[j], false, false);
    }
  }
#pragma unroll
  for (int j = 0; j < 4; ++j)
#pragma unroll
    for (int r = 0; r < 8; ++r)
      C[(size_t)(tm * 16 + r + hi * 8) * N + tn * 64 + j * 16 + r15] = acc[j][r];
}

// ------------------------- RMSNorm + RoPE (fused) --------------------------

__global__ void norm_rope_kernel(const float* __restrict__ proj,
                                 const float* __restrict__ w,
                                 const float* __restrict__ cosc,
                                 const float* __restrict__ sinc,
                                 bf16* __restrict__ out, int nheads) {
  int wid  = (blockIdx.x * blockDim.x + threadIdx.x) >> 5;
  int lane = threadIdx.x & 31;
  if (wid >= S_LEN * nheads) return;
  int s = wid / nheads, h = wid - s * nheads;

  const float* row = proj + ((size_t)s * nheads + h) * HD;
  int j = lane * 2;                       // pairs (j, j+64)
  float a0 = row[j], a1 = row[j + 1];
  float b0 = row[j + 64], b1 = row[j + 65];

  float ss = a0 * a0 + a1 * a1 + b0 * b0 + b1 * b1;
#pragma unroll
  for (int m = 16; m; m >>= 1) ss += __shfl_xor(ss, m, 32);
  float rinv = rsqrtf(ss * (1.0f / 128.0f) + 1e-6f);

  a0 *= rinv * w[j];      a1 *= rinv * w[j + 1];
  b0 *= rinv * w[j + 64]; b1 *= rinv * w[j + 65];

  float c0 = cosc[(size_t)s * 64 + j], c1 = cosc[(size_t)s * 64 + j + 1];
  float s0 = sinc[(size_t)s * 64 + j], s1 = sinc[(size_t)s * 64 + j + 1];

  bf16* orow = out + ((size_t)h * S_LEN + s) * HD;
  orow[j]      = f2bf(a0 * c0 - b0 * s0);
  orow[j + 1]  = f2bf(a1 * c1 - b1 * s1);
  orow[j + 64] = f2bf(b0 * c0 + a0 * s0);
  orow[j + 65] = f2bf(b1 * c1 + a1 * s1);
}

// ---------------------------- flash attention ------------------------------
// Block = 8 waves, one head, 8 consecutive 16-row q-tiles. K/V 32-key chunks
// staged in LDS (TDM double-buffer when available), online softmax per wave.

#define ATT_WAVES 8

__device__ __forceinline__ void attn_wave_chunk(
    const Frag16 qf[4], const bf16* kt, const bf16* vt32, bf16* p,
    int hi, int r15, v8f O[8], float m[8], float l[8]) {
  const float scale = 0.08838834764831845f;   // 1/sqrt(128)
  v8f sc[2];
#pragma unroll
  for (int nt = 0; nt < 2; ++nt) {
    v8f s = {};
    const bf16* kr = kt + (size_t)(nt * 16 + r15) * HD + hi * 16;
#pragma unroll
    for (int c = 0; c < 4; ++c) {
      v16bf b = *(const v16bf*)(kr + c * 32);
      s = __builtin_amdgcn_wmma_f32_16x16x32_bf16(
          false, qf[c].v, false, b, (short)0, s, false, false);
    }
    sc[nt] = s * scale;
  }
#pragma unroll
  for (int r = 0; r < 8; ++r) {
    float t = fmaxf(sc[0][r], sc[1][r]);
#pragma unroll
    for (int msk = 8; msk; msk >>= 1) t = fmaxf(t, __shfl_xor(t, msk, 16));
    float mn = fmaxf(m[r], t);
    float alpha = __expf(m[r] - mn);
    m[r] = mn;
    float p0 = __expf(sc[0][r] - mn);
    float p1 = __expf(sc[1][r] - mn);
    float rs = p0 + p1;
#pragma unroll
    for (int msk = 8; msk; msk >>= 1) rs += __shfl_xor(rs, msk, 16);
    l[r] = l[r] * alpha + rs;
#pragma unroll
    for (int nd = 0; nd < 8; ++nd) O[nd][r] *= alpha;
    int mrow = r + hi * 8;
    p[mrow * 32 + r15]      = f2bf(p0);
    p[mrow * 32 + 16 + r15] = f2bf(p1);
  }
  asm volatile("s_wait_dscnt 0" ::: "memory");   // P store -> P frag reload
  Frag16 pf;
  pf.h[0] = *(const v8bf*)(p + r15 * 32 + hi * 16);
  pf.h[1] = *(const v8bf*)(p + r15 * 32 + 8 + hi * 16);
#pragma unroll
  for (int nd = 0; nd < 8; ++nd) {
    // V tile rows: dim d, 32 keys contiguous
    const bf16* vr = vt32 + (size_t)(nd * 16 + r15) * 32 + hi * 16;
    v16bf b = *(const v16bf*)vr;
    O[nd] = __builtin_amdgcn_wmma_f32_16x16x32_bf16(
        false, pf.v, false, b, (short)0, O[nd], false, false);
  }
}

#if HAVE_TDM
__global__ void attn_kernel_tdm(const bf16* __restrict__ Q,
                                const bf16* __restrict__ Kb,
                                const bf16* __restrict__ Vt,
                                bf16* __restrict__ Oattn) {
  // double buffers: K tile 32x128, V tile 128x32 (bf16) + per-wave P slots
  __shared__ __align__(64) bf16 kbuf[2][32 * HD];
  __shared__ __align__(64) bf16 vbuf[2][HD * 32];
  __shared__ __align__(64) bf16 ldsP[ATT_WAVES][16 * 32];

  int wslot = threadIdx.x >> 5;
  int lane  = threadIdx.x & 31;
  int h     = blockIdx.x >> 5;               // 32 blocks per head
  int tblk  = blockIdx.x & 31;
  int q0    = (tblk * ATT_WAVES + wslot) * 16;
  int kv    = h >> 2;
  int hi = lane >> 4, r15 = lane & 15;

  const bf16* qrow = Q + ((size_t)h * S_LEN + q0 + r15) * HD + hi * 16;
  Frag16 qf[4];
#pragma unroll
  for (int c = 0; c < 4; ++c) {
    qf[c].h[0] = *(const v8bf*)(qrow + c * 32);
    qf[c].h[1] = *(const v8bf*)(qrow + c * 32 + 8);
  }

  const bf16* kbase = Kb + (size_t)kv * S_LEN * HD;
  const bf16* vbase = Vt + (size_t)kv * HD * S_LEN;
  bf16* p = &ldsP[wslot][0];

  v8f O[8] = {};
  float m[8], l[8];
#pragma unroll
  for (int r = 0; r < 8; ++r) { m[r] = -3.0e38f; l[r] = 0.0f; }

  const int nchunks = S_LEN / 32;
  if (wslot == 0) {
    tdm_load_2d(lds_offset(&kbuf[0][0]), kbase, HD, S_LEN, HD, 32, HD);
    tdm_load_2d(lds_offset(&vbuf[0][0]), vbase, S_LEN, HD, 32, HD, S_LEN);
  }
  for (int t = 0; t < nchunks; ++t) {
    int cur = t & 1;
    if (wslot == 0) {
      if (t + 1 < nchunks) {
        int key1 = (t + 1) * 32;
        tdm_load_2d(lds_offset(&kbuf[cur ^ 1][0]), kbase + (size_t)key1 * HD,
                    HD, S_LEN, HD, 32, HD);
        tdm_load_2d(lds_offset(&vbuf[cur ^ 1][0]), vbase + key1,
                    S_LEN, HD, 32, HD, S_LEN);
        __builtin_amdgcn_s_wait_tensorcnt((short)2);   // chunk t landed
      } else {
        __builtin_amdgcn_s_wait_tensorcnt((short)0);
      }
    }
    __syncthreads();                                   // tiles visible to block
    attn_wave_chunk(qf, &kbuf[cur][0], &vbuf[cur][0], p, hi, r15, O, m, l);
    __syncthreads();                                   // done reading buffer
  }

#pragma unroll
  for (int r = 0; r < 8; ++r) {
    float inv = 1.0f / l[r];
    int row = q0 + r + hi * 8;
    bf16* orow = Oattn + (size_t)row * (NHEADS * HD) + h * HD + r15;
#pragma unroll
    for (int nd = 0; nd < 8; ++nd) orow[nd * 16] = f2bf(O[nd][r] * inv);
  }
}
#else
// Fallback: per-wave global-fed flash attention (no TDM builtin available).
__global__ void attn_kernel_global(const bf16* __restrict__ Q,
                                   const bf16* __restrict__ Kb,
                                   const bf16* __restrict__ Vt,
                                   bf16* __restrict__ Oattn) {
  __shared__ __align__(64) bf16 ldsP[ATT_WAVES][16 * 32];
  int wslot = threadIdx.x >> 5;
  int wid   = blockIdx.x * ATT_WAVES + wslot;
  int lane  = threadIdx.x & 31;
  int h     = wid >> 8;
  int tile  = wid & 255;
  int q0    = tile * 16;
  int kv    = h >> 2;
  int hi = lane >> 4, r15 = lane & 15;

  const bf16* qrow = Q + ((size_t)h * S_LEN + q0 + r15) * HD + hi * 16;
  Frag16 qf[4];
#pragma unroll
  for (int c = 0; c < 4; ++c) {
    qf[c].h[0] = *(const v8bf*)(qrow + c * 32);
    qf[c].h[1] = *(const v8bf*)(qrow + c * 32 + 8);
  }
  const bf16* kbase = Kb + (size_t)kv * S_LEN * HD;
  const bf16* vbase = Vt + (size_t)kv * HD * S_LEN;
  bf16* p = &ldsP[wslot][0];

  v8f O[8] = {};
  float m[8], l[8];
#pragma unroll
  for (int r = 0; r < 8; ++r) { m[r] = -3.0e38f; l[r] = 0.0f; }
  const float scale = 0.08838834764831845f;

  for (int key0 = 0; key0 < S_LEN; key0 += 32) {
    v8f sc[2];
#pragma unroll
    for (int nt = 0; nt < 2; ++nt) {
      v8f s = {};
      const bf16* kr = kbase + (size_t)(key0 + nt * 16 + r15) * HD + hi * 16;
#pragma unroll
      for (int c = 0; c < 4; ++c) {
        v16bf b = *(const v16bf*)(kr + c * 32);
        s = __builtin_amdgcn_wmma_f32_16x16x32_bf16(
            false, qf[c].v, false, b, (short)0, s, false, false);
      }
      sc[nt] = s * scale;
    }
#pragma unroll
    for (int r = 0; r < 8; ++r) {
      float t = fmaxf(sc[0][r], sc[1][r]);
#pragma unroll
      for (int msk = 8; msk; msk >>= 1) t = fmaxf(t, __shfl_xor(t, msk, 16));
      float mn = fmaxf(m[r], t);
      float alpha = __expf(m[r] - mn);
      m[r] = mn;
      float p0 = __expf(sc[0][r] - mn);
      float p1 = __expf(sc[1][r] - mn);
      float rs = p0 + p1;
#pragma unroll
      for (int msk = 8; msk; msk >>= 1) rs += __shfl_xor(rs, msk, 16);
      l[r] = l[r] * alpha + rs;
#pragma unroll
      for (int nd = 0; nd < 8; ++nd) O[nd][r] *= alpha;
      int mrow = r + hi * 8;
      p[mrow * 32 + r15]      = f2bf(p0);
      p[mrow * 32 + 16 + r15] = f2bf(p1);
    }
    asm volatile("s_wait_dscnt 0" ::: "memory");
    Frag16 pf;
    pf.h[0] = *(const v8bf*)(p + r15 * 32 + hi * 16);
    pf.h[1] = *(const v8bf*)(p + r15 * 32 + 8 + hi * 16);
#pragma unroll
    for (int nd = 0; nd < 8; ++nd) {
      const bf16* vr = vbase + (size_t)(nd * 16 + r15) * S_LEN + key0 + hi * 16;
      v16bf b = *(const v16bf*)vr;
      O[nd] = __builtin_amdgcn_wmma_f32_16x16x32_bf16(
          false, pf.v, false, b, (short)0, O[nd], false, false);
    }
  }
#pragma unroll
  for (int r = 0; r < 8; ++r) {
    float inv = 1.0f / l[r];
    int row = q0 + r + hi * 8;
    bf16* orow = Oattn + (size_t)row * (NHEADS * HD) + h * HD + r15;
#pragma unroll
    for (int nd = 0; nd < 8; ++nd) orow[nd * 16] = f2bf(O[nd][r] * inv);
  }
}
#endif

// ------------------------------- launcher ----------------------------------

extern "C" void kernel_launch(void* const* d_in, const int* in_sizes, int n_in,
                              void* d_out, int out_size, void* d_ws, size_t ws_size,
                              hipStream_t stream) {
  const float* x    = (const float*)d_in[0];
  const float* Wq   = (const float*)d_in[1];
  const float* Wk   = (const float*)d_in[2];
  const float* Wv   = (const float*)d_in[3];
  const float* Wo   = (const float*)d_in[4];
  const float* qnw  = (const float*)d_in[5];
  const float* knw  = (const float*)d_in[6];
  const float* cosc = (const float*)d_in[7];
  const float* sinc = (const float*)d_in[8];
  float* out = (float*)d_out;

  char* w = (char*)d_ws;
  const size_t MB = (size_t)1 << 20;
  bf16*  xb  = (bf16*) (w + 0 * MB);    // x bf16                16 MB
  bf16*  wqt = (bf16*) (w + 16 * MB);   // Wq^T bf16              8 MB
  bf16*  wkt = (bf16*) (w + 24 * MB);   // Wk^T bf16              2 MB
  bf16*  wvt = (bf16*) (w + 26 * MB);   // Wv^T bf16              2 MB
  bf16*  wot = (bf16*) (w + 28 * MB);   // Wo^T bf16              8 MB
  float* qp  = (float*)(w + 36 * MB);   // q proj f32            32 MB
  float* kp  = (float*)(w + 68 * MB);   // k proj f32             8 MB
  float* vp  = (float*)(w + 76 * MB);   // v proj f32             8 MB
  bf16*  qb  = (bf16*) (w + 84 * MB);   // q bf16 [H][S][HD]     16 MB
  bf16*  kb  = (bf16*) (w + 100 * MB);  // k bf16 [KV][S][HD]     4 MB
  bf16*  vt  = (bf16*) (w + 104 * MB);  // v bf16 [KV][HD][S]     4 MB
  bf16*  ab  = (bf16*) (w + 108 * MB);  // attn out bf16         16 MB

  const int T = 256;

  cast_bf16_kernel<<<(S_LEN * DMODEL + T - 1) / T, T, 0, stream>>>(x, xb, S_LEN * DMODEL);
  cast_transpose_kernel<<<(DMODEL * DMODEL + T - 1) / T, T, 0, stream>>>(Wq, wqt, DMODEL, DMODEL);
  cast_transpose_kernel<<<(DMODEL * NKV * HD + T - 1) / T, T, 0, stream>>>(Wk, wkt, DMODEL, NKV * HD);
  cast_transpose_kernel<<<(DMODEL * NKV * HD + T - 1) / T, T, 0, stream>>>(Wv, wvt, DMODEL, NKV * HD);
  cast_transpose_kernel<<<(DMODEL * DMODEL + T - 1) / T, T, 0, stream>>>(Wo, wot, DMODEL, DMODEL);

  {
    int waves_q = (S_LEN / 16) * (DMODEL / 64);
    gemm_bf16_kernel<<<waves_q / 8, T, 0, stream>>>(xb, wqt, qp, S_LEN, DMODEL, DMODEL);
    int waves_kv = (S_LEN / 16) * ((NKV * HD) / 64);
    gemm_bf16_kernel<<<waves_kv / 8, T, 0, stream>>>(xb, wkt, kp, S_LEN, NKV * HD, DMODEL);
    gemm_bf16_kernel<<<waves_kv / 8, T, 0, stream>>>(xb, wvt, vp, S_LEN, NKV * HD, DMODEL);
  }

  norm_rope_kernel<<<(S_LEN * NHEADS) / 8, T, 0, stream>>>(qp, qnw, cosc, sinc, qb, NHEADS);
  norm_rope_kernel<<<(S_LEN * NKV) / 8, T, 0, stream>>>(kp, knw, cosc, sinc, kb, NKV);
  v_transpose_kernel<<<(S_LEN * NKV * HD + T - 1) / T, T, 0, stream>>>(vp, vt);

#if HAVE_TDM
  attn_kernel_tdm<<<NHEADS * (S_LEN / 16 / ATT_WAVES), T, 0, stream>>>(qb, kb, vt, ab);
#else
  attn_kernel_global<<<(NHEADS * (S_LEN / 16)) / ATT_WAVES, T, 0, stream>>>(qb, kb, vt, ab);
#endif

  {
    int waves_o = (S_LEN / 16) * (DMODEL / 64);
    gemm_bf16_kernel<<<waves_o / 8, T, 0, stream>>>(ab, wot, out, S_LEN, DMODEL, DMODEL);
  }
}
